// KGencoder_17660905521751
// MI455X (gfx1250) — compile-verified, matrix operand order checked by reading, not run on hardware
//
#include <hip/hip_runtime.h>
#include <string.h>

// ---------------------------------------------------------------------------
// RGCN (KGencoder) for MI455X / gfx1250, wave32 + WMMA bf16.
//   out = [Hmean | x] @ [W ; root]  + bias + x
// ---------------------------------------------------------------------------

typedef __attribute__((ext_vector_type(16))) __bf16 v16bf;
typedef __attribute__((ext_vector_type(8)))  float  v8f;

#define NNODE   100000
#define RREL    8
#define DDIM    128
#define KTOT    1152                 // R*D + D
#define NKT     (KTOT / 32)          // 36 k-steps of the bf16 WMMA
#define ROWTILES (NNODE / 16)        // 6250 (exact)
#define BPACK_DWORDS (NKT * 8 * 32 * 8)   // 73728 dwords = 288 KB

// K index (low element of the bf16 pair) held by (lane-half, vgpr v) for the
// 16-bit A/B fragment layout (ISA 7.12.2, "16-bit A-Matrix 16x32"):
//   lanes 0-15 : v0..3 -> K=0..7,   v4..7 -> K=16..23
//   lanes 16-31: v0..3 -> K=8..15,  v4..7 -> K=24..31
__device__ __forceinline__ int kmap(int half, int v) {
    return (v < 4) ? ((half ? 8 : 0) + 2 * v)
                   : ((half ? 24 : 16) + 2 * (v - 4));
}

// --------------------------- kernel 1: zero H/cnt ---------------------------
__global__ void zero_ws(float4* __restrict__ h4, long nh4,
                        float4* __restrict__ c4, long nc4) {
    long i = (long)blockIdx.x * blockDim.x + threadIdx.x;
    long stride = (long)gridDim.x * blockDim.x;
    float4 z = {0.f, 0.f, 0.f, 0.f};
    for (long j = i; j < nh4; j += stride) h4[j] = z;
    for (long j = i; j < nc4; j += stride) c4[j] = z;
}

// ------------------- kernel 2: pack [W;root] -> B fragments -----------------
// Bpack dword index = ((kt*8 + ct)*32 + lane)*8 + v ; each dword = bf16 pair
// (K, K+1) of column n = ct*16 + (lane&15).
__global__ void pack_B(const float* __restrict__ weight,  // [1024][128]
                       const float* __restrict__ root,    // [128][128]
                       unsigned int* __restrict__ bpack) {
    int idx = blockIdx.x * blockDim.x + threadIdx.x;
    if (idx >= BPACK_DWORDS) return;
    int v    = idx & 7;
    int lane = (idx >> 3) & 31;
    int ct   = (idx >> 8) & 7;
    int kt   = idx >> 11;
    int half = lane >> 4;
    int n    = ct * 16 + (lane & 15);
    int k    = kt * 32 + kmap(half, v);
    float b0, b1;
    if (k < RREL * DDIM) {   // weight region: flat [r*128+d][e] == [k][n]
        b0 = weight[(size_t)k * DDIM + n];
        b1 = weight[(size_t)(k + 1) * DDIM + n];
    } else {                 // root region
        b0 = root[(size_t)(k - 1024) * DDIM + n];
        b1 = root[(size_t)(k - 1023) * DDIM + n];
    }
    union { __bf16 b[2]; unsigned int u; } pk;
    pk.b[0] = (__bf16)b0;
    pk.b[1] = (__bf16)b1;
    bpack[idx] = pk.u;
}

// ------------------- kernel 3: edge scatter (segment sums) ------------------
// One wave per edge; lane covers 4 of the 128 features.
__global__ void scatter_edges(const float* __restrict__ x,
                              const int*   __restrict__ ei,   // [2][E]
                              const int*   __restrict__ et,   // [E]
                              float* __restrict__ H,
                              float* __restrict__ cnt, int E) {
    int wid  = (int)((blockIdx.x * blockDim.x + threadIdx.x) >> 5);
    int lane = threadIdx.x & 31;
    if (wid >= E) return;
    int src = ei[wid];
    int dst = ei[E + wid];
    int typ = et[wid];
    const float4 m = *(const float4*)(x + (size_t)src * DDIM + lane * 4);
    float* hseg = H + ((size_t)dst * RREL + typ) * DDIM + lane * 4;
    unsafeAtomicAdd(hseg + 0, m.x);   // global_atomic_add_f32 (no return)
    unsafeAtomicAdd(hseg + 1, m.y);
    unsafeAtomicAdd(hseg + 2, m.z);
    unsafeAtomicAdd(hseg + 3, m.w);
    if (lane == 0) unsafeAtomicAdd(cnt + (size_t)dst * RREL + typ, 1.0f);
}

// ------------------- kernel 4: fused mean + GEMM + epilogue -----------------
// 256 threads = 8 waves. Whole packed B lives in LDS (288 KB <= 320 KB/WGP).
// Each wave: one 16-row strip x all 128 cols -> 8 wmma accumulators.
// K-loop: outer r = relation block (0..7 = Hmean with 1/cnt scale, 8 = x with
// scale 1.0 feeding the root projection), inner 4 k-steps of 32.
__global__ void rgcn_gemm(const float* __restrict__ H,
                          const float* __restrict__ cnt,
                          const float* __restrict__ x,
                          const float* __restrict__ bias,
                          const unsigned int* __restrict__ bpack,
                          float* __restrict__ out) {
    extern __shared__ unsigned int smem[];   // BPACK_DWORDS dwords

    // cooperative linear copy of packed B into LDS
    const float4* gsrc = (const float4*)bpack;
    float4*       ldst = (float4*)smem;
    for (int i = threadIdx.x; i < BPACK_DWORDS / 4; i += blockDim.x)
        ldst[i] = gsrc[i];
    __syncthreads();

    int wave = threadIdx.x >> 5;
    int lane = threadIdx.x & 31;
    int rt   = blockIdx.x * 8 + wave;
    if (rt >= ROWTILES) return;

    int half = lane >> 4;
    int arow = rt * 16 + (lane & 15);      // A-fragment: lane holds row M

    v8f acc[8];
#pragma unroll
    for (int ct = 0; ct < 8; ++ct)
        acc[ct] = (v8f){0.f,0.f,0.f,0.f,0.f,0.f,0.f,0.f};

    const float* hrow = H + (size_t)arow * (RREL * DDIM);
    const float* xrow = x + (size_t)arow * DDIM;

    // per-wave base of this lane's B-fragment dwords in LDS
    const float4* bsm = (const float4*)(smem + (size_t)lane * 8);

    for (int r = 0; r < 9; ++r) {
        float        s    = (r < 8) ? 1.0f / fmaxf(cnt[(size_t)arow*RREL + r], 1.0f)
                                    : 1.0f;
        const float* base = (r < 8) ? (hrow + r * DDIM) : xrow;
#pragma unroll
        for (int q = 0; q < 4; ++q) {
            int kt = r * 4 + q;
            const float* aptr = base + q * 32;

            // build bf16 A fragment (scaled mean / raw x)
            v16bf afrag;
#pragma unroll
            for (int v = 0; v < 8; ++v) {
                float2 hv = *(const float2*)(aptr + kmap(half, v));
                afrag[2 * v]     = (__bf16)(hv.x * s);
                afrag[2 * v + 1] = (__bf16)(hv.y * s);
            }

            // preload all 8 B fragments as float4 pairs -> ds_load_b128 x2
            union BFrag { float4 f[2]; v16bf v; };
            BFrag bfr[8];
#pragma unroll
            for (int ct = 0; ct < 8; ++ct) {
                // fragment dword offset = ((kt*8+ct)*32 + lane)*8 ; /4 -> f4 idx
                const float4* p = bsm + (size_t)(kt * 8 + ct) * 64;
                bfr[ct].f[0] = p[0];
                bfr[ct].f[1] = p[1];
            }

            // 8 back-to-back WMMAs (independent accumulators, shared A)
#pragma unroll
            for (int ct = 0; ct < 8; ++ct)
                acc[ct] = __builtin_amdgcn_wmma_f32_16x16x32_bf16(
                    false, afrag, false, bfr[ct].v, (short)0, acc[ct],
                    false, false);
        }
    }

    // epilogue: C/D layout -> lane = N (per half), vgpr v = M (v + 8*half)
#pragma unroll
    for (int ct = 0; ct < 8; ++ct) {
        int   col = ct * 16 + (lane & 15);
        float b   = bias[col];
#pragma unroll
        for (int v = 0; v < 8; ++v) {
            int    row = rt * 16 + 8 * half + v;
            size_t o   = (size_t)row * DDIM + col;
            out[o] = acc[ct][v] + b + x[o];   // + residual
        }
    }
}

// ---------------------------------------------------------------------------
extern "C" void kernel_launch(void* const* d_in, const int* in_sizes, int n_in,
                              void* d_out, int out_size, void* d_ws, size_t ws_size,
                              hipStream_t stream) {
    const float* x      = (const float*)d_in[0];   // [N,128]
    const float* weight = (const float*)d_in[1];   // [8,128,128]
    const float* root   = (const float*)d_in[2];   // [128,128]
    const float* bias   = (const float*)d_in[3];   // [128]
    const int*   eidx   = (const int*)d_in[4];     // [2,E]
    const int*   etyp   = (const int*)d_in[5];     // [E]
    float*       out    = (float*)d_out;
    const int E = in_sizes[5];

    // workspace layout (16B aligned offsets)
    char*  ws    = (char*)d_ws;
    float* H     = (float*)ws;                                      // 409,600,000 B
    float* cnt   = (float*)(ws + (size_t)NNODE * RREL * DDIM * 4);  //   3,200,000 B
    unsigned int* bpack =
        (unsigned int*)(ws + (size_t)NNODE * RREL * DDIM * 4
                           + (size_t)NNODE * RREL * 4);             //     294,912 B

    // 1) zero segment sums + counts
    zero_ws<<<2048, 256, 0, stream>>>(
        (float4*)H,   (long)((size_t)NNODE * RREL * DDIM / 4),
        (float4*)cnt, (long)((size_t)NNODE * RREL / 4));

    // 2) pack [W ; root] into bf16 WMMA B-fragment order
    pack_B<<<(BPACK_DWORDS + 255) / 256, 256, 0, stream>>>(weight, root, bpack);

    // 3) atomic scatter of edge messages (one wave per edge)
    scatter_edges<<<(E + 7) / 8, 256, 0, stream>>>(x, eidx, etyp, H, cnt, E);

    // 4) fused mean + bf16 WMMA GEMM + bias + residual
    int smem_bytes = BPACK_DWORDS * 4;   // 294,912 B of LDS (<= 320 KB/WGP)
    (void)hipFuncSetAttribute((const void*)rgcn_gemm,
                              hipFuncAttributeMaxDynamicSharedMemorySize,
                              smem_bytes);
    int blocks = (ROWTILES + 7) / 8;     // 782
    rgcn_gemm<<<blocks, 256, smem_bytes, stream>>>(H, cnt, x, bias, bpack, out);
}
